// FeedforwardClusterNet_20014547599878
// MI455X (gfx1250) — matrix operand order repaired and checked by reading, not compiled, not run on hardware
//
#include <hip/hip_runtime.h>
#include <cstddef>
#include <cstdint>

// ---------------------------------------------------------------------------
// Types for CDNA5 WMMA (gfx1250, wave32)
// ---------------------------------------------------------------------------
typedef __bf16 bf16;
typedef __attribute__((ext_vector_type(16))) __bf16 v16bf;
typedef __attribute__((ext_vector_type(8)))  float  v8f;
typedef __attribute__((ext_vector_type(4)))  int    v4i;
typedef __attribute__((address_space(1))) v4i gas_v4i;  // global (LangAS __device__)
typedef __attribute__((address_space(3))) v4i las_v4i;  // LDS    (LangAS __shared__)

#define CLUSTER_TEMP 5.0f

// Async global->LDS DMA (gfx1250). Guarded: falls back to b128 LDS stores.
#if defined(__has_builtin)
#if __has_builtin(__builtin_amdgcn_global_load_async_to_lds_b128) && \
    __has_builtin(__builtin_amdgcn_s_wait_asynccnt)
#define HAVE_ASYNC_LDS 1
#endif
#endif
#ifndef HAVE_ASYNC_LDS
#define HAVE_ASYNC_LDS 0
#endif

__device__ __forceinline__ void copy16_g2l(const bf16* g, bf16* l) {
#if HAVE_ASYNC_LDS
  __builtin_amdgcn_global_load_async_to_lds_b128((gas_v4i*)g, (las_v4i*)l, 0, 0);
#else
  *reinterpret_cast<uint4*>(l) = *reinterpret_cast<const uint4*>(g);
#endif
}

__device__ __forceinline__ void wait_async_then_barrier() {
#if HAVE_ASYNC_LDS
  __builtin_amdgcn_s_wait_asynccnt(0);
#endif
  __syncthreads();
}

// ---------------------------------------------------------------------------
// Fragment loaders (ISA 7.12.2 layouts)
//  A (16x32 bf16): lane L -> row = m0 + L%16; two 8-elem K runs at
//                  kb = k0 + 8*(L/16) and kb+16  (each run = 16B)
//  B (32x16 bf16): lane L -> col = n0 + L%16; two 8-elem K runs at
//                  kb = k0 + 8*(L/16) and kb+16  (each run = 16B)
//  C/D (16x16 f32): vgpr r -> row = 8*(L/16)+r, col = L%16
// ---------------------------------------------------------------------------
__device__ __forceinline__ v16bf frag_a(const bf16* base, int lda, int m0, int k0, int lane) {
  int row = m0 + (lane & 15);
  int kb  = k0 + ((lane >> 4) << 3);
  const bf16* p = base + (size_t)row * lda + kb;
  v16bf a;
  uint4* ap = reinterpret_cast<uint4*>(&a);
  ap[0] = *reinterpret_cast<const uint4*>(p);
  ap[1] = *reinterpret_cast<const uint4*>(p + 16);
  return a;
}

__device__ __forceinline__ v16bf frag_b(const bf16* baseT, int lda, int n0, int k0, int lane) {
  int col = n0 + (lane & 15);
  int kb  = k0 + ((lane >> 4) << 3);
  const bf16* p = baseT + (size_t)col * lda + kb;
  v16bf b;
  uint4* bp = reinterpret_cast<uint4*>(&b);
  bp[0] = *reinterpret_cast<const uint4*>(p);
  bp[1] = *reinterpret_cast<const uint4*>(p + 16);
  return b;
}

__device__ __forceinline__ v8f wmma_bf16(v16bf a, v16bf b, v8f c) {
  return __builtin_amdgcn_wmma_f32_16x16x32_bf16(false, a, false, b, (short)0, c, false, false);
}

// ---------------------------------------------------------------------------
// Tiled bf16 GEMM:  C[M,N] = act(A[M,K] @ Bt[N,K]^T + bias)
//  - Bt is the PRE-TRANSPOSED weight matrix [N][K] so both LDS tiles are
//    staged with pure 16B copies (async DMA when available).
//  - Block tile 128(M) x 128(N), 8 waves as 4x2, wave tile 32x64,
//    K-step 32, double-buffered LDS: DMA of tile s+1 overlaps WMMA on tile s.
// ---------------------------------------------------------------------------
#define GP 40  // LDS pitch (elements) for 32-wide K tiles (bank-conflict pad)

template<bool RELU, bool OUT_BF16>
__global__ __launch_bounds__(256)
void gemm_bias_kernel(const bf16* __restrict__ A, const bf16* __restrict__ Bt,
                      const float* __restrict__ bias, void* __restrict__ Cout,
                      int M, int K, int N) {
  __shared__ bf16 As[2][128 * GP];
  __shared__ bf16 Bs[2][128 * GP];
  const int tid  = threadIdx.x;
  const int lane = tid & 31;
  const int wave = tid >> 5;
  const int mw   = wave >> 1;   // 0..3 -> rows 32*mw
  const int nw   = wave & 1;    // 0..1 -> cols 64*nw
  const int m_blk = blockIdx.x * 128;
  const int n_blk = blockIdx.y * 128;

  v8f acc[2][4];
#pragma unroll
  for (int h = 0; h < 2; ++h)
#pragma unroll
    for (int t = 0; t < 4; ++t)
#pragma unroll
      for (int i = 0; i < 8; ++i) acc[h][t][i] = 0.0f;

  auto stage = [&](int buf, int k0) {
    // A tile: 128 rows x 32 k = 512 x 16B chunks (2 per thread), rows clamped.
    for (int c = tid; c < 512; c += 256) {
      int row = c >> 2, kk = (c & 3) << 3;
      int gr = m_blk + row;
      if (gr >= M) gr = M - 1;  // clamp (dup data; stores guarded later)
      copy16_g2l(A + (size_t)gr * K + k0 + kk, &As[buf][row * GP + kk]);
    }
    // B tile from pre-transposed weights: 128 n-rows x 32 k, always in range.
    for (int c = tid; c < 512; c += 256) {
      int row = c >> 2, kk = (c & 3) << 3;
      copy16_g2l(Bt + (size_t)(n_blk + row) * K + k0 + kk, &Bs[buf][row * GP + kk]);
    }
  };

  auto compute = [&](int buf) {
    v16bf a0 = frag_a(As[buf], GP, mw * 32,      0, lane);
    v16bf a1 = frag_a(As[buf], GP, mw * 32 + 16, 0, lane);
#pragma unroll
    for (int t = 0; t < 4; ++t) {
      v16bf b = frag_b(Bs[buf], GP, nw * 64 + t * 16, 0, lane);
      acc[0][t] = wmma_bf16(a0, b, acc[0][t]);
      acc[1][t] = wmma_bf16(a1, b, acc[1][t]);
    }
  };

  const int nsteps = K >> 5;
  stage(0, 0);
  wait_async_then_barrier();
  for (int s = 0; s < nsteps; ++s) {
    int buf = s & 1;
    if (s + 1 < nsteps) stage(buf ^ 1, (s + 1) << 5);  // DMA next tile
    compute(buf);                                       // WMMA current tile
    wait_async_then_barrier();
  }

#pragma unroll
  for (int h = 0; h < 2; ++h) {
    int row0 = m_blk + mw * 32 + h * 16 + ((lane >> 4) << 3);
#pragma unroll
    for (int t = 0; t < 4; ++t) {
      int col  = n_blk + nw * 64 + t * 16 + (lane & 15);
      float bv = bias[col];
#pragma unroll
      for (int rr = 0; rr < 8; ++rr) {
        int row = row0 + rr;
        if (row < M) {
          float v = acc[h][t][rr] + bv;
          if (RELU) v = fmaxf(v, 0.0f);
          if (OUT_BF16) ((bf16*)Cout)[(size_t)row * N + col] = (bf16)v;
          else          ((float*)Cout)[(size_t)row * N + col] = v;
        }
      }
    }
  }
}

// ---------------------------------------------------------------------------
// Fused soft-k-means step:
//  dist = data @ mu^T (WMMA, mu LDS-resident = native Bt layout),
//  softmax(5*dist) in registers (shfl within 16-lane halves),
//  mean += r^T @ data (WMMA on rT / dataT LDS tiles) + cluster_r.
// ---------------------------------------------------------------------------
#define MU_PITCH 264
#define DT_PITCH 136
#define RT_PITCH 136
#define SMEM_MU_BYTES (64 * MU_PITCH * 2)
#define SMEM_DT_OFF   SMEM_MU_BYTES
#define SMEM_DT_BYTES (256 * DT_PITCH * 2)
#define SMEM_RT_OFF   (SMEM_DT_OFF + SMEM_DT_BYTES)
#define SMEM_RT_BYTES (64 * RT_PITCH * 2)
#define SMEM_TOTAL    (SMEM_RT_OFF + SMEM_RT_BYTES)   // 120832 B -> 2 WGs/WGP

template<bool FINAL>
__global__ __launch_bounds__(256)
void cluster_step_kernel(const bf16* __restrict__ data, const bf16* __restrict__ mu,
                         float* __restrict__ mean_acc, float* __restrict__ cr_acc,
                         float* __restrict__ r_out, int n) {
  extern __shared__ char smem[];
  bf16* mu_s = (bf16*)smem;
  bf16* dT_s = (bf16*)(smem + SMEM_DT_OFF);
  bf16* rT_s = (bf16*)(smem + SMEM_RT_OFF);
  const int tid  = threadIdx.x;
  const int lane = tid & 31;
  const int wave = tid >> 5;
  const int m_blk = blockIdx.x * 128;

  for (int e = tid; e < 64 * 256; e += 256)
    mu_s[(e >> 8) * MU_PITCH + (e & 255)] = mu[e];

  if (!FINAL) {
    for (int e = tid; e < 128 * 256; e += 256) {
      int row = e >> 8, d = e & 255;
      int gr = m_blk + row;
      dT_s[d * DT_PITCH + row] = (gr < n) ? data[(size_t)gr * 256 + d] : (bf16)0.0f;
    }
  }
  __syncthreads();

  // ---- GEMM-A: dist[128,64] = data @ mu^T; wave w owns rows [16w,16w+16)
  int grow = m_blk + wave * 16 + (lane & 15);
  if (grow >= n) grow = n - 1;
  const bf16* arow = data + (size_t)grow * 256 + ((lane >> 4) << 3);

  v8f acc[4];
#pragma unroll
  for (int t = 0; t < 4; ++t)
#pragma unroll
    for (int i = 0; i < 8; ++i) acc[t][i] = 0.0f;

#pragma unroll
  for (int k0 = 0; k0 < 256; k0 += 32) {
    v16bf a;
    uint4* ap = reinterpret_cast<uint4*>(&a);
    ap[0] = *reinterpret_cast<const uint4*>(arow + k0);
    ap[1] = *reinterpret_cast<const uint4*>(arow + k0 + 16);
#pragma unroll
    for (int t = 0; t < 4; ++t) {
      v16bf b = frag_b(mu_s, MU_PITCH, t * 16, k0, lane);
      acc[t] = wmma_bf16(a, b, acc[t]);
    }
  }

  // ---- softmax over 64 clusters; a row lives in one 16-lane half at fixed rr
#pragma unroll
  for (int rr = 0; rr < 8; ++rr) {
    float m = acc[0][rr];
#pragma unroll
    for (int t = 1; t < 4; ++t) m = fmaxf(m, acc[t][rr]);
#pragma unroll
    for (int sh = 1; sh < 16; sh <<= 1) m = fmaxf(m, __shfl_xor(m, sh, 32));
    float ex[4], s = 0.0f;
#pragma unroll
    for (int t = 0; t < 4; ++t) { ex[t] = __expf(CLUSTER_TEMP * (acc[t][rr] - m)); s += ex[t]; }
#pragma unroll
    for (int sh = 1; sh < 16; sh <<= 1) s += __shfl_xor(s, sh, 32);
    float inv = 1.0f / s;
#pragma unroll
    for (int t = 0; t < 4; ++t) acc[t][rr] = ex[t] * inv;
  }

  if (FINAL) {
#pragma unroll
    for (int t = 0; t < 4; ++t) {
      int col = t * 16 + (lane & 15);
#pragma unroll
      for (int rr = 0; rr < 8; ++rr) {
        int row = m_blk + wave * 16 + ((lane >> 4) << 3) + rr;
        if (row < n) r_out[(size_t)row * 64 + col] = acc[t][rr];
      }
    }
    return;
  }

  // ---- r^T tile -> LDS bf16 (A operand of the mean GEMM); zero padded rows
#pragma unroll
  for (int t = 0; t < 4; ++t) {
    int c = t * 16 + (lane & 15);
#pragma unroll
    for (int rr = 0; rr < 8; ++rr) {
      int rl = wave * 16 + ((lane >> 4) << 3) + rr;
      float v = (m_blk + rl < n) ? acc[t][rr] : 0.0f;
      rT_s[c * RT_PITCH + rl] = (bf16)v;
    }
  }
  __syncthreads();

  if (tid < 64) {
    float s = 0.0f;
    for (int i = 0; i < 128; ++i) s += (float)rT_s[tid * RT_PITCH + i];
    atomicAdd(&cr_acc[tid], s);
  }

  // ---- GEMM-B: mean[64,256] += rT[64,128] @ data[128,256]
  const int mt = (wave & 3) * 16;
#pragma unroll
  for (int half = 0; half < 2; ++half) {
    int nb = ((wave >> 2) + 2 * half) * 64;
    v8f acc2[4];
#pragma unroll
    for (int t = 0; t < 4; ++t)
#pragma unroll
      for (int i = 0; i < 8; ++i) acc2[t][i] = 0.0f;
#pragma unroll
    for (int k0 = 0; k0 < 128; k0 += 32) {
      v16bf a = frag_a(rT_s, RT_PITCH, mt, k0, lane);
#pragma unroll
      for (int t = 0; t < 4; ++t) {
        v16bf b = frag_b(dT_s, DT_PITCH, nb + t * 16, k0, lane);
        acc2[t] = wmma_bf16(a, b, acc2[t]);
      }
    }
    int c0 = mt + ((lane >> 4) << 3);
#pragma unroll
    for (int t = 0; t < 4; ++t) {
      int col = nb + t * 16 + (lane & 15);
#pragma unroll
      for (int rr = 0; rr < 8; ++rr)
        atomicAdd(&mean_acc[(c0 + rr) * 256 + col], acc2[t][rr]);
    }
  }
}

// ---------------------------------------------------------------------------
// Small helper kernels
// ---------------------------------------------------------------------------
__global__ void cvt_f32_to_bf16_kernel(const float* __restrict__ in, bf16* __restrict__ out, size_t cnt) {
  size_t i = (size_t)blockIdx.x * blockDim.x + threadIdx.x;
  size_t stride = (size_t)gridDim.x * blockDim.x;
  for (; i < cnt; i += stride) out[i] = (bf16)in[i];
}

// W[K][N] f32  ->  Wt[N][K] bf16 (one-time; makes GEMM B-staging pure b128)
__global__ void cvt_transpose_kernel(const float* __restrict__ in, bf16* __restrict__ out,
                                     int K, int N) {
  size_t cnt = (size_t)K * N;
  size_t i = (size_t)blockIdx.x * blockDim.x + threadIdx.x;
  size_t stride = (size_t)gridDim.x * blockDim.x;
  for (; i < cnt; i += stride) {
    int k = (int)(i / N), nn = (int)(i % N);
    out[(size_t)nn * K + k] = (bf16)in[i];
  }
}

__global__ void zero_f32_kernel(float* __restrict__ p, int cnt) {
  int i = blockIdx.x * blockDim.x + threadIdx.x;
  if (i < cnt) p[i] = 0.0f;
}

// One wave per row: L2-normalize embeds[n,256] and emit bf16.
__global__ void normalize_rows_kernel(const float* __restrict__ emb, bf16* __restrict__ out, int n) {
  int wave = threadIdx.x >> 5, lane = threadIdx.x & 31;
  int row = blockIdx.x * 8 + wave;
  if (row >= n) return;
  const float* e = emb + (size_t)row * 256;
  float v[8], s = 0.0f;
#pragma unroll
  for (int i = 0; i < 8; ++i) { v[i] = e[lane + 32 * i]; s += v[i] * v[i]; }
#pragma unroll
  for (int sh = 16; sh >= 1; sh >>= 1) s += __shfl_xor(s, sh, 32);
  float inv = rsqrtf(s);
#pragma unroll
  for (int i = 0; i < 8; ++i) out[(size_t)row * 256 + lane + 32 * i] = (bf16)(v[i] * inv);
}

__global__ void update_mu_kernel(const float* __restrict__ mean_acc, const float* __restrict__ cr,
                                 bf16* __restrict__ mu_b) {
  int i = blockIdx.x * 256 + threadIdx.x;
  if (i < 64 * 256) mu_b[i] = (bf16)(mean_acc[i] / cr[i >> 8]);
}

// ---------------------------------------------------------------------------
// Host orchestration (capture-safe: kernels only, all on `stream`)
// ---------------------------------------------------------------------------
extern "C" void kernel_launch(void* const* d_in, const int* in_sizes, int n_in,
                              void* d_out, int out_size, void* d_ws, size_t ws_size,
                              hipStream_t stream) {
  const float* x   = (const float*)d_in[0];
  const float* W1  = (const float*)d_in[1];
  const float* b1  = (const float*)d_in[2];
  const float* W2  = (const float*)d_in[3];
  const float* b2  = (const float*)d_in[4];
  const float* W3  = (const float*)d_in[5];
  const float* b3  = (const float*)d_in[6];
  const float* mu0 = (const float*)d_in[7];
  const int n = in_sizes[0] / 512;       // 50000
  const int NUM_ITER = 10;               // fixed by setup_inputs

  char* ws = (char*)d_ws;
  size_t o = 0;
  auto alloc = [&](size_t bytes) { size_t r = o; o += (bytes + 255) & ~(size_t)255; return r; };
  size_t o_xb   = alloc((size_t)n * 512 * 2);     // x bf16 ... later emb f32 (same bytes)
  size_t o_w1   = alloc((size_t)512 * 1024 * 2);  // W1^T [1024][512]
  size_t o_w2   = alloc((size_t)1024 * 1024 * 2); // W2^T [1024][1024]
  size_t o_w3   = alloc((size_t)1024 * 256 * 2);  // W3^T [256][1024]
  size_t o_h1   = alloc((size_t)n * 1024 * 2);    // h1 bf16 ... later data bf16
  size_t o_h2   = alloc((size_t)n * 1024 * 2);
  size_t o_mub  = alloc((size_t)64 * 256 * 2);
  size_t o_mean = alloc((size_t)64 * 256 * 4);    // cr immediately follows
  size_t o_cr   = alloc((size_t)64 * 4);
  (void)ws_size; (void)n_in; (void)out_size;

  bf16*  xb    = (bf16*)(ws + o_xb);
  bf16*  w1t   = (bf16*)(ws + o_w1);
  bf16*  w2t   = (bf16*)(ws + o_w2);
  bf16*  w3t   = (bf16*)(ws + o_w3);
  bf16*  h1b   = (bf16*)(ws + o_h1);
  bf16*  h2b   = (bf16*)(ws + o_h2);
  float* emb   = (float*)(ws + o_xb);   // reuse
  bf16*  datab = (bf16*)(ws + o_h1);    // reuse
  bf16*  mub   = (bf16*)(ws + o_mub);
  float* meanp = (float*)(ws + o_mean);
  float* crp   = (float*)(ws + o_cr);

  cvt_f32_to_bf16_kernel<<<(int)(((size_t)n * 512 + 1023) / 1024), 1024, 0, stream>>>(
      x, xb, (size_t)n * 512);
  cvt_f32_to_bf16_kernel<<<16, 1024, 0, stream>>>(mu0, mub, (size_t)64 * 256);
  cvt_transpose_kernel<<<512, 1024, 0, stream>>>(W1, w1t, 512, 1024);
  cvt_transpose_kernel<<<1024, 1024, 0, stream>>>(W2, w2t, 1024, 1024);
  cvt_transpose_kernel<<<256, 1024, 0, stream>>>(W3, w3t, 1024, 256);

  const int mtiles = (n + 127) / 128;
  dim3 g1(mtiles, 1024 / 128);
  gemm_bias_kernel<true, true><<<g1, 256, 0, stream>>>(xb,  w1t, b1, (void*)h1b, n, 512,  1024);
  gemm_bias_kernel<true, true><<<g1, 256, 0, stream>>>(h1b, w2t, b2, (void*)h2b, n, 1024, 1024);
  dim3 g3(mtiles, 256 / 128);
  gemm_bias_kernel<false, false><<<g3, 256, 0, stream>>>(h2b, w3t, b3, (void*)emb, n, 1024, 256);

  normalize_rows_kernel<<<(n + 7) / 8, 256, 0, stream>>>(emb, datab, n);

  (void)hipFuncSetAttribute((const void*)cluster_step_kernel<false>,
                            hipFuncAttributeMaxDynamicSharedMemorySize, SMEM_TOTAL);
  (void)hipFuncSetAttribute((const void*)cluster_step_kernel<true>,
                            hipFuncAttributeMaxDynamicSharedMemorySize, SMEM_TOTAL);

  for (int it = 0; it < NUM_ITER; ++it) {
    zero_f32_kernel<<<(16448 + 255) / 256, 256, 0, stream>>>(meanp, 16448);
    cluster_step_kernel<false><<<mtiles, 256, SMEM_TOTAL, stream>>>(datab, mub, meanp, crp, nullptr, n);
    update_mu_kernel<<<64, 256, 0, stream>>>(meanp, crp, mub);
  }
  cluster_step_kernel<true><<<mtiles, 256, SMEM_TOTAL, stream>>>(datab, mub, nullptr, nullptr,
                                                                 (float*)d_out, n);
}
